// ZSGGNN_21251498180706
// MI455X (gfx1250) — compile-verified, matrix operand order checked by reading, not run on hardware
//
#include <hip/hip_runtime.h>

// ZSGGNN full pipeline for gfx1250 (MI455X, wave32, WMMA).
// Big GEMMs: bf16 WMMA 16x16x32 with f32 accumulate.
//   - Block tile 128x64, BK=32; each wave: 2 A-frags x 2 B-frags = 4 WMMAs/step.
//   - LDS tiles in fragment-ready swizzled layout (ds_load_b128 fragment reads).
//   - Clamped staging addresses; no row/col value masks (OOB rows/cols only feed
//     never-stored C entries); K-mask applied only on the tail K-block.
//   - Software pipelined: next tile's global loads stay in flight across WMMA.
// Small ops: wave32 shuffle-reduction kernels.

typedef __bf16 bf16_t;
typedef __attribute__((ext_vector_type(16))) __bf16 v16bf;
typedef __attribute__((ext_vector_type(4)))  __bf16 v4bf;
typedef __attribute__((ext_vector_type(8)))  float  v8f;

#define B_    16
#define L_    196
#define FD_   2048
#define N_    1000
#define WD_   300
#define HD_   512
#define DD_   5
#define E_    50000
#define NIMG_ 1000

__device__ __forceinline__ bf16_t f2bf(float f) { return (bf16_t)f; }

// ---------------------------------------------------------------------------
// Tiled WMMA GEMM: C[M,N] = act(A[M,K] @ B + bias)
// ACT: 0=none 1=relu 2=tanh ; TRANSB: B stored [N,K] ; GATHER: A row mg is
// concat(embed[e0[mg]], embed[e1[mg]]) (edge features, K=600).
// m0: global row offset of this launch (C indexed by local row).
// ---------------------------------------------------------------------------
template<int ACT, bool TRANSB, bool GATHER>
__global__ __launch_bounds__(256) void k_gemm(
    int M, int N, int K, int m0,
    const float* __restrict__ A, int lda, long long sA,
    const float* __restrict__ Bm, int ldb, long long sB,
    float* __restrict__ C, int ldc, long long sC,
    const float* __restrict__ bias,
    const float* __restrict__ embed,
    const int* __restrict__ ed0,
    const int* __restrict__ ed1) {
  // Fragment-ready swizzle: slot = (subtile*32 + wmma_lane)*16 + elem, 32B/lane.
  __shared__ __align__(32) bf16_t Asw[8 * 32 * 16];   // 8 row-subtiles of 16x32
  __shared__ __align__(32) bf16_t Bsw[4 * 32 * 16];   // 4 col-subtiles of 32x16
  const int tid  = threadIdx.x;
  const int lane = tid & 31, wid = tid >> 5;
  const int tr = wid >> 1, tc = wid & 1;      // wave: rows tr*32+.., cols tc*32+..
  const int z = blockIdx.z;
  const float* Ab = A  + (size_t)z * sA;
  const float* Bb = Bm + (size_t)z * sB;
  float*       Cb = C  + (size_t)z * sC;
  const int mBase = blockIdx.y * 128;
  const int n0    = blockIdx.x * 64;

  // ---- loop-invariant staging descriptors (k-quads per thread) ----
  int    aK[4], aIdx[4], aN0[4], aN1[4];
  size_t aOff[4];
  #pragma unroll
  for (int i = 0; i < 4; ++i) {
    int q = i * 256 + tid;
    int r = q >> 3, kq = (q & 7) * 4;               // row 0..127, kquad 0..28
    int st = r >> 4, m = r & 15;
    int g = kq >> 4, rem = kq & 15, h2 = rem >> 3, w0 = rem & 7;
    aK[i]   = kq;
    aIdx[i] = ((st * 32 + (m + h2 * 16)) << 4) + g * 8 + w0;
    int mg = m0 + mBase + r;
    int mgc = mg < M ? mg : (M - 1);                // clamp: OOB rows feed only
    aOff[i] = (size_t)mgc * lda;                    // never-stored C rows
    if (GATHER) { aN0[i] = ed0[mgc]; aN1[i] = ed1[mgc]; }
    else        { aN0[i] = 0;        aN1[i] = 0;        }
  }
  int    bK[2], bIdx[2];
  size_t bOff[2];
  #pragma unroll
  for (int i = 0; i < 2; ++i) {
    int q = i * 256 + tid;
    int nl = q & 63, kq = (q >> 6) * 4;             // col 0..63, kquad 0..28
    int sn = nl >> 4, nn = nl & 15;
    int g = kq >> 4, rem = kq & 15, h2 = rem >> 3, w0 = rem & 7;
    bK[i]   = kq;
    bIdx[i] = ((sn * 32 + (nn + h2 * 16)) << 4) + g * 8 + w0;
    int ng = n0 + nl;
    int ngc = ng < N ? ng : (N - 1);                // clamp: OOB cols harmless
    bOff[i] = TRANSB ? (size_t)ngc * ldb : (size_t)ngc;
  }

  // tile load into registers; k-mask only on tail block (uniform `tail`)
  auto loadTiles = [&](int k0, bool tail, float (&ra)[4][4], float (&rb)[2][4]) {
    #pragma unroll
    for (int i = 0; i < 4; ++i) {
      #pragma unroll
      for (int j = 0; j < 4; ++j) {
        int kg = k0 + aK[i] + j;
        int kgc = kg < K ? kg : (K - 1);
        float v;
        if (GATHER) {
          int node = (kgc < WD_) ? aN0[i] : aN1[i];
          int cc   = (kgc < WD_) ? kgc : kgc - WD_;
          v = embed[(size_t)node * WD_ + cc];
        } else {
          v = Ab[aOff[i] + kgc];
        }
        if (tail && kg >= K) v = 0.f;
        ra[i][j] = v;
      }
    }
    #pragma unroll
    for (int i = 0; i < 2; ++i) {
      #pragma unroll
      for (int j = 0; j < 4; ++j) {
        int kg = k0 + bK[i] + j;
        int kgc = kg < K ? kg : (K - 1);
        float v = TRANSB ? Bb[bOff[i] + kgc] : Bb[(size_t)kgc * ldb + bOff[i]];
        if (tail && kg >= K) v = 0.f;
        rb[i][j] = v;
      }
    }
  };
  // convert + one ds_store_b64 per quad
  auto storeTiles = [&](const float (&ra)[4][4], const float (&rb)[2][4]) {
    #pragma unroll
    for (int i = 0; i < 4; ++i) {
      v4bf a;
      #pragma unroll
      for (int j = 0; j < 4; ++j) a[j] = f2bf(ra[i][j]);
      *(v4bf*)&Asw[aIdx[i]] = a;
    }
    #pragma unroll
    for (int i = 0; i < 2; ++i) {
      v4bf b;
      #pragma unroll
      for (int j = 0; j < 4; ++j) b[j] = f2bf(rb[i][j]);
      *(v4bf*)&Bsw[bIdx[i]] = b;
    }
  };

  v8f acc00 = {}, acc01 = {}, acc10 = {}, acc11 = {};
  float ra[4][4], rb[2][4];
  loadTiles(0, 32 > K, ra, rb);
  storeTiles(ra, rb);
  __syncthreads();

  for (int k0 = 0; k0 < K; k0 += 32) {
    const int  kn   = k0 + 32;
    const bool more = kn < K;
    if (more) loadTiles(kn, (kn + 32) > K, ra, rb);  // in flight across WMMA
    const v16bf* Afr = (const v16bf*)Asw;
    const v16bf* Bfr = (const v16bf*)Bsw;
    v16bf a0 = Afr[(tr * 2 + 0) * 32 + lane];
    v16bf a1 = Afr[(tr * 2 + 1) * 32 + lane];
    v16bf b0 = Bfr[(tc * 2 + 0) * 32 + lane];
    v16bf b1 = Bfr[(tc * 2 + 1) * 32 + lane];
    acc00 = __builtin_amdgcn_wmma_f32_16x16x32_bf16(false, a0, false, b0,
                                                    (short)0, acc00, false, false);
    acc01 = __builtin_amdgcn_wmma_f32_16x16x32_bf16(false, a0, false, b1,
                                                    (short)0, acc01, false, false);
    acc10 = __builtin_amdgcn_wmma_f32_16x16x32_bf16(false, a1, false, b0,
                                                    (short)0, acc10, false, false);
    acc11 = __builtin_amdgcn_wmma_f32_16x16x32_bf16(false, a1, false, b1,
                                                    (short)0, acc11, false, false);
    __syncthreads();
    if (more) { storeTiles(ra, rb); __syncthreads(); }
  }

  // epilogue: bias + activation + guarded store
  #pragma unroll
  for (int i = 0; i < 2; ++i) {
    #pragma unroll
    for (int j = 0; j < 2; ++j) {
      v8f acc = (i == 0) ? ((j == 0) ? acc00 : acc01)
                         : ((j == 0) ? acc10 : acc11);
      int nl = n0 + tc * 32 + j * 16 + (lane & 15);
      if (nl >= N) continue;
      #pragma unroll
      for (int r = 0; r < 8; ++r) {
        int ml = mBase + tr * 32 + i * 16 + r + 8 * (lane >> 4);
        int mg = m0 + ml;
        if (mg < M) {
          float v = acc[r];
          if (bias) v += bias[nl];
          if (ACT == 1) v = v > 0.f ? v : 0.f;
          else if (ACT == 2) v = tanhf(v);
          Cb[(size_t)ml * ldc + nl] = v;
        }
      }
    }
  }
}

// ---------------------------------------------------------------------------
// mean-pool over L + imagenet classifier + softmax (one block per batch row)
// ---------------------------------------------------------------------------
__global__ void k_pool_clf(const float* __restrict__ feats,
                           const float* __restrict__ Wc,
                           const float* __restrict__ bc,
                           float* __restrict__ out) {
  __shared__ float pooled[FD_];
  __shared__ float red[256];
  const int b = blockIdx.x, tid = threadIdx.x;
  for (int f = tid; f < FD_; f += 256) {
    float s = 0.f;
    for (int l = 0; l < L_; ++l) s += feats[((size_t)b * L_ + l) * FD_ + f];
    pooled[f] = s * (1.0f / L_);
  }
  __syncthreads();
  float lg[4];
  #pragma unroll
  for (int i = 0; i < 4; ++i) {
    int n = tid + i * 256;
    float a = -1e30f;
    if (n < NIMG_) {
      a = bc[n];
      for (int f = 0; f < FD_; ++f) a += pooled[f] * Wc[(size_t)f * NIMG_ + n];
    }
    lg[i] = a;
  }
  float mx = fmaxf(fmaxf(lg[0], lg[1]), fmaxf(lg[2], lg[3]));
  red[tid] = mx; __syncthreads();
  for (int s = 128; s > 0; s >>= 1) {
    if (tid < s) red[tid] = fmaxf(red[tid], red[tid + s]);
    __syncthreads();
  }
  mx = red[0]; __syncthreads();
  float ssum = 0.f;
  #pragma unroll
  for (int i = 0; i < 4; ++i) {
    int n = tid + i * 256;
    if (n < NIMG_) { lg[i] = __expf(lg[i] - mx); ssum += lg[i]; }
  }
  red[tid] = ssum; __syncthreads();
  for (int s = 128; s > 0; s >>= 1) {
    if (tid < s) red[tid] += red[tid + s];
    __syncthreads();
  }
  float inv = 1.0f / red[0];
  #pragma unroll
  for (int i = 0; i < 4; ++i) {
    int n = tid + i * 256;
    if (n < NIMG_) out[(size_t)b * NIMG_ + n] = lg[i] * inv;
  }
}

// row-wise softmax, one wave32 per row
__global__ void k_softmax_rows(float* __restrict__ p, int rows, int cols) {
  const int wid = threadIdx.x >> 5, lane = threadIdx.x & 31;
  const int row = blockIdx.x * 8 + wid;
  if (row >= rows) return;
  float* r = p + (size_t)row * cols;
  float mx = -1e30f;
  for (int c = lane; c < cols; c += 32) mx = fmaxf(mx, r[c]);
  for (int o = 16; o > 0; o >>= 1) mx = fmaxf(mx, __shfl_xor(mx, o, 32));
  float s = 0.f;
  for (int c = lane; c < cols; c += 32) { float e = __expf(r[c] - mx); r[c] = e; s += e; }
  for (int o = 16; o > 0; o >>= 1) s += __shfl_xor(s, o, 32);
  float inv = 1.0f / s;
  for (int c = lane; c < cols; c += 32) r[c] *= inv;
}

__global__ void k_zero(float* __restrict__ p, int n) {
  int i = blockIdx.x * 256 + threadIdx.x;
  if (i < n) p[i] = 0.f;
}

// edge value = tanh(hidden256 . Wr2 + br2), scatter-add into prop_mat. 1 wave/edge.
__global__ void k_edge_vals(const float* __restrict__ eh,
                            const float* __restrict__ Wr2,
                            const float* __restrict__ br2,
                            const int* __restrict__ ed0, const int* __restrict__ ed1,
                            int start, int count, float* __restrict__ prop) {
  const int wid = threadIdx.x >> 5, lane = threadIdx.x & 31;
  const int li = blockIdx.x * 8 + wid;
  if (li >= count) return;
  float acc = 0.f;
  for (int j = lane; j < 256; j += 32) acc += eh[(size_t)li * 256 + j] * Wr2[j];
  for (int o = 16; o > 0; o >>= 1) acc += __shfl_xor(acc, o, 32);
  if (lane == 0) {
    float v = tanhf(acc + br2[0]);
    int e = start + li;
    atomicAdd(&prop[(size_t)ed0[e] * N_ + ed1[e]], v);
  }
}

// msg = tanh(prop_mat @ hidden) fused with GRU cell. One block per (n, b).
__global__ void k_msg_gru(const float* __restrict__ prop,
                          const float* __restrict__ hin, float* __restrict__ hout,
                          const float* __restrict__ Wih, const float* __restrict__ Whh,
                          const float* __restrict__ bih, const float* __restrict__ bhh) {
  const int n = blockIdx.x, b = blockIdx.y, tid = threadIdx.x;
  const int lane = tid & 31, wid = tid >> 5;
  float acc[5] = {0.f, 0.f, 0.f, 0.f, 0.f};
  const float* hb = hin + (size_t)b * N_ * DD_;
  const float* pr = prop + (size_t)n * N_;
  for (int k = tid; k < N_; k += 256) {
    float p = pr[k];
    #pragma unroll
    for (int d = 0; d < 5; ++d) acc[d] += p * hb[(size_t)k * DD_ + d];
  }
  #pragma unroll
  for (int d = 0; d < 5; ++d)
    for (int o = 16; o > 0; o >>= 1) acc[d] += __shfl_xor(acc[d], o, 32);
  __shared__ float red[8][5];
  if (lane == 0)
    for (int d = 0; d < 5; ++d) red[wid][d] = acc[d];
  __syncthreads();
  if (tid == 0) {
    float x[5], h[5];
    for (int d = 0; d < 5; ++d) {
      float m = 0.f;
      for (int w = 0; w < 8; ++w) m += red[w][d];
      x[d] = tanhf(m);
      h[d] = hin[((size_t)b * N_ + n) * DD_ + d];
    }
    float gi[15], gh[15];
    for (int j = 0; j < 15; ++j) {
      float a = bih[j], c2 = bhh[j];
      for (int d = 0; d < 5; ++d) { a += x[d] * Wih[j * DD_ + d]; c2 += h[d] * Whh[j * DD_ + d]; }
      gi[j] = a; gh[j] = c2;
    }
    for (int d = 0; d < 5; ++d) {
      float r  = 1.f / (1.f + __expf(-(gi[d] + gh[d])));
      float zg = 1.f / (1.f + __expf(-(gi[5 + d] + gh[5 + d])));
      float ng = tanhf(gi[10 + d] + r * gh[10 + d]);
      hout[((size_t)b * N_ + n) * DD_ + d] = (1.f - zg) * ng + zg * h[d];
    }
  }
}

// logits = relu(h @ Wo1 + bo1) @ Wo2 + bo2.  One wave32 per (b,n) row.
__global__ void k_out_mlp(const float* __restrict__ h,
                          const float* __restrict__ Wo1, const float* __restrict__ bo1,
                          const float* __restrict__ Wo2, const float* __restrict__ bo2,
                          float* __restrict__ logits) {
  const int wid = threadIdx.x >> 5, lane = threadIdx.x & 31;
  const int row = blockIdx.x * 8 + wid;
  if (row >= B_ * N_) return;
  float hv[5];
  #pragma unroll
  for (int d = 0; d < 5; ++d) hv[d] = h[(size_t)row * DD_ + d];
  float acc = 0.f;
  for (int j = lane; j < 512; j += 32) {
    float t = bo1[j];
    #pragma unroll
    for (int d = 0; d < 5; ++d) t += hv[d] * Wo1[d * 512 + j];
    t = t > 0.f ? t : 0.f;
    acc += t * Wo2[j];
  }
  for (int o = 16; o > 0; o >>= 1) acc += __shfl_xor(acc, o, 32);
  if (lane == 0) logits[row] = acc + bo2[0];
}

// ---------------------------------------------------------------------------
extern "C" void kernel_launch(void* const* d_in, const int* in_sizes, int n_in,
                              void* d_out, int out_size, void* d_ws, size_t ws_size,
                              hipStream_t stream) {
  const float* feats = (const float*)d_in[0];
  const float* embed = (const float*)d_in[1];
  const int*   edges = (const int*)d_in[2];
  const float* Wkey  = (const float*)d_in[3];
  const float* bkey  = (const float*)d_in[4];
  const float* Wq    = (const float*)d_in[5];
  const float* bq    = (const float*)d_in[6];
  const float* Wf1   = (const float*)d_in[7];
  const float* bf1   = (const float*)d_in[8];
  const float* Wf2   = (const float*)d_in[9];
  const float* bf2   = (const float*)d_in[10];
  const float* Wr1   = (const float*)d_in[11];
  const float* br1   = (const float*)d_in[12];
  const float* Wr2   = (const float*)d_in[13];
  const float* br2   = (const float*)d_in[14];
  const float* Wo1   = (const float*)d_in[15];
  const float* bo1   = (const float*)d_in[16];
  const float* Wo2   = (const float*)d_in[17];
  const float* bo2   = (const float*)d_in[18];
  const float* Wih   = (const float*)d_in[19];
  const float* bih   = (const float*)d_in[20];
  const float* Whh   = (const float*)d_in[21];
  const float* bhh   = (const float*)d_in[22];
  const float* Wclf  = (const float*)d_in[23];
  const float* bclf  = (const float*)d_in[24];
  (void)in_sizes; (void)n_in; (void)out_size; (void)ws_size;

  float* out    = (float*)d_out;
  float* logits = out;                                   // [16,1000]
  float* attn   = out + (size_t)B_ * N_;                 // [16,1000,196]
  float* imgnet = attn + (size_t)B_ * N_ * L_;           // [16,1000]

  float* ws = (float*)d_ws;
  size_t o = 0;
  float* keys = ws + o; o += (size_t)(B_ * L_) * HD_;    // 3136x512
  float* qrs  = ws + o; o += (size_t)N_ * HD_;           // 1000x512
  float* prop = ws + o; o += (size_t)N_ * N_;            // 1000x1000
  float* hA   = ws + o; o += (size_t)B_ * N_ * DD_;      // 16000x5
  float* hB   = ws + o; o += (size_t)B_ * N_ * DD_;      // 16000x5
  float* gi   = ws + o; o += (size_t)N_ * FD_;           // 1000x2048 (per-b chunk)
  float* h1   = ws + o; o += (size_t)8192 * 256;         // max(1000x512, 8192x256)

  const dim3 blk(256);
  const float* nulf = nullptr;
  const int*   nuli = nullptr;

  // 1) imagenet head
  k_pool_clf<<<dim3(B_), blk, 0, stream>>>(feats, Wclf, bclf, imgnet);

  // 2) keys = feats @ W_key + b_key   [3136 x 512]
  k_gemm<0, false, false><<<dim3(8, 25, 1), blk, 0, stream>>>(
      B_ * L_, HD_, FD_, 0, feats, FD_, 0, Wkey, HD_, 0,
      keys, HD_, 0, bkey, nulf, nuli, nuli);

  // 3) queries = embed @ W_query + b_query   [1000 x 512]
  k_gemm<0, false, false><<<dim3(8, 8, 1), blk, 0, stream>>>(
      N_, HD_, WD_, 0, embed, WD_, 0, Wq, HD_, 0,
      qrs, HD_, 0, bq, nulf, nuli, nuli);

  // 4) scores[b] = queries @ keys[b]^T   batched over z, into attn region
  k_gemm<0, true, false><<<dim3(4, 8, B_), blk, 0, stream>>>(
      N_, L_, HD_, 0, qrs, HD_, 0, keys, HD_, (long long)L_ * HD_,
      attn, L_, (long long)N_ * L_, nulf, nulf, nuli, nuli);

  // 5) softmax over L
  k_softmax_rows<<<dim3((B_ * N_ + 7) / 8), blk, 0, stream>>>(attn, B_ * N_, L_);

  // 6) per-batch: graph_input -> relu MLP -> h0
  for (int b = 0; b < B_; ++b) {
    k_gemm<0, false, false><<<dim3(32, 8, 1), blk, 0, stream>>>(
        N_, FD_, L_, 0, attn + (size_t)b * N_ * L_, L_, 0,
        feats + (size_t)b * L_ * FD_, FD_, 0,
        gi, FD_, 0, nulf, nulf, nuli, nuli);
    k_gemm<1, false, false><<<dim3(8, 8, 1), blk, 0, stream>>>(
        N_, 512, FD_, 0, gi, FD_, 0, Wf1, 512, 0,
        h1, 512, 0, bf1, nulf, nuli, nuli);
    k_gemm<0, false, false><<<dim3(1, 8, 1), blk, 0, stream>>>(
        N_, DD_, 512, 0, h1, 512, 0, Wf2, DD_, 0,
        hA + (size_t)b * N_ * DD_, DD_, 0, bf2, nulf, nuli, nuli);
  }

  // 7) prop_mat: zero, then edge MLP (gathered A) + tanh + scatter-add, chunked
  k_zero<<<dim3((N_ * N_ + 255) / 256), blk, 0, stream>>>(prop, N_ * N_);
  const int CH = 8192;
  for (int c = 0; c * CH < E_; ++c) {
    int start = c * CH;
    int cnt = (E_ - start < CH) ? (E_ - start) : CH;
    k_gemm<1, false, true><<<dim3(4, (cnt + 127) / 128, 1), blk, 0, stream>>>(
        E_, 256, 2 * WD_, start, nulf, 2 * WD_, 0, Wr1, 256, 0,
        h1, 256, 0, br1, embed, edges, edges + E_);
    k_edge_vals<<<dim3((cnt + 7) / 8), blk, 0, stream>>>(
        h1, Wr2, br2, edges, edges + E_, start, cnt, prop);
  }

  // 8) 5 message-passing + GRU steps (ping-pong buffers)
  float* hc = hA; float* hn = hB;
  for (int t = 0; t < 5; ++t) {
    k_msg_gru<<<dim3(N_, B_), blk, 0, stream>>>(prop, hc, hn, Wih, Whh, bih, bhh);
    float* tmp = hc; hc = hn; hn = tmp;
  }

  // 9) output MLP -> logits
  k_out_mlp<<<dim3((B_ * N_ + 7) / 8), blk, 0, stream>>>(hc, Wo1, bo1, Wo2, bo2, logits);
}
